// MoE_90847148245561
// MI455X (gfx1250) — compile-verified, matrix operand order checked by reading, not run on hardware
//
#include <hip/hip_runtime.h>
#include <hip/hip_bf16.h>

typedef __attribute__((ext_vector_type(16))) __bf16 v16bf;
typedef __attribute__((ext_vector_type(8)))  __bf16 bf16x8;
typedef __attribute__((ext_vector_type(4)))  __bf16 bf16x4;
typedef __attribute__((ext_vector_type(8)))  float  v8f;
typedef __attribute__((ext_vector_type(4)))  unsigned int u32x4;
typedef __attribute__((ext_vector_type(8)))  int i32x8;
typedef __attribute__((ext_vector_type(4)))  int i32x4;

#define E_ 8
#define B_ 4096
#define D_ 2048
#define T_ 2048
#define H_ 2048

// Block = 256 threads = 8 waves (2 M x 4 N), wave tile 64x32, block tile 128x128.
#define BM 128
#define BN 128
#define BK 32

#if __has_builtin(__builtin_amdgcn_tensor_load_to_lds) && \
    __has_builtin(__builtin_amdgcn_s_wait_tensorcnt)
#define USE_TDM 1
#else
#define USE_TDM 0
#endif

// ---------------------------------------------------------------------------
// Fragment load. Per CDNA5 ISA 16-bit WMMA layout, lane L of an A (16x32) or
// B (32x16) bf16 fragment owns row/col (L%16) and K values in two contiguous
// 8-element runs: K in [kh, kh+8) and [kh+16, kh+24), kh = 8*(L/16).
// With A stored [m][k] and B stored transposed [n][k] in LDS, a fragment is
// exactly two ds_load_b128.
// ---------------------------------------------------------------------------
__device__ inline v16bf frag_ld(const __bf16* __restrict__ rowbase, int kh) {
  bf16x8 lo = *(const bf16x8*)(rowbase + kh);
  bf16x8 hi = *(const bf16x8*)(rowbase + kh + 16);
  return __builtin_shufflevector(lo, hi, 0, 1, 2, 3, 4, 5, 6, 7,
                                 8, 9, 10, 11, 12, 13, 14, 15);
}

#if USE_TDM
// ---------------------------------------------------------------------------
// Tensor Data Mover: load a 2D tile (tile_w elems of 2B x tile_h rows, row
// stride row_stride elems) from global into LDS at byte offset lds_off.
// D# per CDNA5 ISA (count=1, type=2 "image",
// data_size=1 -> 2 bytes, tensor dims sized >= tile so no OOB clipping).
// 6-arg builtin form (clang-23 / therock-10.0 headers).
// ---------------------------------------------------------------------------
__device__ inline void tdm_load_tile_2d(unsigned lds_off, const void* gaddr,
                                        unsigned tile_w, unsigned tile_h,
                                        unsigned row_stride,
                                        unsigned tensor_w, unsigned tensor_h) {
  unsigned long long ga = (unsigned long long)(uintptr_t)gaddr;
  u32x4 g0;
  g0[0] = 1u;                                           // count=1, user D#
  g0[1] = lds_off;                                      // lds_addr (bytes)
  g0[2] = (unsigned)(ga & 0xFFFFFFFFu);                 // global_addr[31:0]
  g0[3] = (unsigned)((ga >> 32) & 0x01FFFFFFu)          // global_addr[56:32]
          | (2u << 30);                                 // type=2
  i32x8 g1;
  g1[0] = (int)(1u << 16);                              // data_size=1 (2B)
  g1[1] = (int)((tensor_w & 0xFFFFu) << 16);            // tensor_dim0[15:0]
  g1[2] = (int)((tensor_w >> 16) | ((tensor_h & 0xFFFFu) << 16));
  g1[3] = (int)((tensor_h >> 16) | (tile_w << 16));     // tile_dim0
  g1[4] = (int)(tile_h & 0xFFFFu);                      // tile_dim1, tile_dim2=0
  g1[5] = (int)row_stride;                              // tensor_dim0_stride lo
  g1[6] = 0;
  g1[7] = 0;
  i32x4 z4 = {0, 0, 0, 0};                              // groups 2/3 unused (2D)
  i32x8 z8 = {0, 0, 0, 0, 0, 0, 0, 0};
  __builtin_amdgcn_tensor_load_to_lds(g0, g1, z4, z4, z8, 0);
}
#endif

// ---------------------------------------------------------------------------
__global__ __launch_bounds__(256) void zero_kernel(float4* __restrict__ out4,
                                                   int* __restrict__ counts) {
  size_t i = (size_t)blockIdx.x * 256 + threadIdx.x;
  size_t stride = (size_t)gridDim.x * 256;
  float4 z = make_float4(0.f, 0.f, 0.f, 0.f);
  for (size_t j = i; j < (size_t)B_ * T_ / 4; j += stride) out4[j] = z;
  if (i < E_) counts[i] = 0;
}

// ---------------------------------------------------------------------------
// Gating: one wave32 per token; softmax over E=8; top-2 (lowest index wins
// ties, matching jax.lax.top_k); append token to selected experts' lists.
// ---------------------------------------------------------------------------
__global__ __launch_bounds__(256) void gate_topk_kernel(
    const float* __restrict__ xs, const float* __restrict__ Wg,
    const float* __restrict__ bg, float* __restrict__ topv_out,
    int* __restrict__ counts, int* __restrict__ tok_idx,
    float* __restrict__ gates) {
  int lane = threadIdx.x & 31;
  int b = blockIdx.x * 8 + (threadIdx.x >> 5);
  if (b >= B_) return;

  float acc[E_];
#pragma unroll
  for (int e = 0; e < E_; ++e) acc[e] = 0.0f;

  for (int d = lane; d < D_; d += 32) {
    float w = Wg[d];
#pragma unroll
    for (int e = 0; e < E_; ++e)
      acc[e] += xs[(size_t)e * B_ * D_ + (size_t)b * D_ + d] * w;
  }
#pragma unroll
  for (int e = 0; e < E_; ++e) {
#pragma unroll
    for (int off = 16; off > 0; off >>= 1)
      acc[e] += __shfl_xor(acc[e], off, 32);
  }

  if (lane == 0) {
    float bgv = bg[0];
    float mx = -1e30f;
#pragma unroll
    for (int e = 0; e < E_; ++e) { acc[e] += bgv; mx = fmaxf(mx, acc[e]); }
    float sum = 0.0f;
#pragma unroll
    for (int e = 0; e < E_; ++e) { acc[e] = __expf(acc[e] - mx); sum += acc[e]; }
    float inv = 1.0f / sum;
#pragma unroll
    for (int e = 0; e < E_; ++e) acc[e] *= inv;

    int i0 = 0; float p0 = acc[0];
#pragma unroll
    for (int e = 1; e < E_; ++e) if (acc[e] > p0) { p0 = acc[e]; i0 = e; }
    int i1 = -1; float p1 = -1.0f;
#pragma unroll
    for (int e = 0; e < E_; ++e) if (e != i0 && acc[e] > p1) { p1 = acc[e]; i1 = e; }

    topv_out[b * 2 + 0] = p0;
    topv_out[b * 2 + 1] = p1;

    int r0 = atomicAdd(&counts[i0], 1);
    tok_idx[i0 * B_ + r0] = b;
    gates[i0 * B_ + r0] = p0;
    int r1 = atomicAdd(&counts[i1], 1);
    tok_idx[i1 * B_ + r1] = b;
    gates[i1 * B_ + r1] = p1;
  }
}

// ---------------------------------------------------------------------------
// GEMM1: hidden[e][r][:] = relu( xs[e][tok[r]][:] @ W1[e] + b1[e] ) in bf16.
// A: gathered token rows, float4 loads -> packed bf16x4 LDS stores, [m][k].
// B: k-strided coalesced loads -> bf16x8 LDS stores, transposed [n][k].
// ---------------------------------------------------------------------------
__global__ __launch_bounds__(256) void expert_ffn1_kernel(
    const float* __restrict__ xs, const float* __restrict__ W1,
    const float* __restrict__ b1, const int* __restrict__ counts,
    const int* __restrict__ tok_idx, __bf16* __restrict__ hidden) {
  int e = blockIdx.z;
  int n_e = counts[e];
  int row0 = blockIdx.y * BM;
  if (row0 >= n_e) return;
  int col0 = blockIdx.x * BN;

  __shared__ __bf16 As[BM][BK];
  __shared__ __bf16 Bs[BN][BK];  // transposed: [n][k]
  __shared__ int tokS[BM];

  int tid = threadIdx.x;
  if (tid < BM) {
    int r = row0 + tid;
    tokS[tid] = (r < n_e) ? tok_idx[e * B_ + r] : 0;
  }
  __syncthreads();

  // A staging: 4 float4 chunks per thread, fixed (m, kq) across the K loop.
  int mA[4], kqA[4];
  const float* aPtr[4];
#pragma unroll
  for (int i = 0; i < 4; ++i) {
    int c = tid + i * 256;            // 1024 chunks = 128 rows x 8 quads
    mA[i] = c >> 3;
    kqA[i] = (c & 7) << 2;
    aPtr[i] = xs + (size_t)e * B_ * D_ + (size_t)tokS[mA[i]] * D_ + kqA[i];
  }
  // B staging: fixed column n, 16-long k segment per thread.
  int nB = tid & 127;
  int ksg = (tid >> 7) << 4;          // 0 or 16
  const float* bPtr = W1 + (size_t)e * D_ * H_ + (size_t)ksg * H_ + col0 + nB;

  int wave = tid >> 5, lane = tid & 31;
  int wm = (wave >> 2) * 64;          // 0 / 64
  int wn = (wave & 3) * 32;           // 0..96
  int lrow = lane & 15;
  int kh = (lane >> 4) << 3;          // 0 or 8

  v8f acc[4][2] = {};

  for (int k0 = 0; k0 < D_; k0 += BK) {
    float4 av[4];
#pragma unroll
    for (int i = 0; i < 4; ++i) av[i] = *(const float4*)(aPtr[i] + k0);
    float bv[16];
#pragma unroll
    for (int kk = 0; kk < 16; ++kk) bv[kk] = bPtr[(size_t)(k0 + kk) * H_];

    __syncthreads();  // previous iteration's fragment reads complete
#pragma unroll
    for (int i = 0; i < 4; ++i) {
      bf16x4 p = { (__bf16)av[i].x, (__bf16)av[i].y,
                   (__bf16)av[i].z, (__bf16)av[i].w };
      *(bf16x4*)(&As[mA[i]][kqA[i]]) = p;
    }
    bf16x8 p0, p1;
#pragma unroll
    for (int kk = 0; kk < 8; ++kk) {
      p0[kk] = (__bf16)bv[kk];
      p1[kk] = (__bf16)bv[kk + 8];
    }
    *(bf16x8*)(&Bs[nB][ksg]) = p0;
    *(bf16x8*)(&Bs[nB][ksg + 8]) = p1;
    __syncthreads();

    v16bf bf0 = frag_ld(&Bs[wn + lrow][0], kh);
    v16bf bf1 = frag_ld(&Bs[wn + 16 + lrow][0], kh);
#pragma unroll
    for (int fm = 0; fm < 4; ++fm) {
      v16bf af = frag_ld(&As[wm + fm * 16 + lrow][0], kh);
      acc[fm][0] = __builtin_amdgcn_wmma_f32_16x16x32_bf16(
          false, af, false, bf0, (short)0, acc[fm][0], false, false);
      acc[fm][1] = __builtin_amdgcn_wmma_f32_16x16x32_bf16(
          false, af, false, bf1, (short)0, acc[fm][1], false, false);
    }
  }

  // Epilogue: D layout -> N = lane%16, M = i + 8*(lane/16).
  int mo = (lane >> 4) << 3;
  const float* b1e = b1 + e * H_;
#pragma unroll
  for (int fm = 0; fm < 4; ++fm) {
#pragma unroll
    for (int fn = 0; fn < 2; ++fn) {
#pragma unroll
      for (int i = 0; i < 8; ++i) {
        int rloc = wm + fm * 16 + mo + i;
        int h = col0 + wn + fn * 16 + lrow;
        float v = fmaxf(acc[fm][fn][i] + b1e[h], 0.0f);
        hidden[((size_t)e * B_ + row0 + rloc) * H_ + h] = (__bf16)v;
      }
    }
  }
}

// ---------------------------------------------------------------------------
// GEMM2: y = hidden[e][r] @ W2[e]; out[tok[r]][t] += gate[r]*(y + b2[e][t]).
// A (already bf16, contiguous 128x64B tile, row stride H*2B) is staged by the
// Tensor Data Mover when available; B converts fp32->bf16 as in GEMM1.
// ---------------------------------------------------------------------------
__global__ __launch_bounds__(256) void expert_ffn2_kernel(
    const __bf16* __restrict__ hidden, const float* __restrict__ W2,
    const float* __restrict__ b2, const int* __restrict__ counts,
    const int* __restrict__ tok_idx, const float* __restrict__ gates,
    float* __restrict__ out) {
  int e = blockIdx.z;
  int n_e = counts[e];
  int row0 = blockIdx.y * BM;
  if (row0 >= n_e) return;
  int col0 = blockIdx.x * BN;

  __shared__ __bf16 As[BM][BK];
  __shared__ __bf16 Bs[BN][BK];  // transposed: [n][k]
  __shared__ int tokS[BM];
  __shared__ float gS[BM];

  int tid = threadIdx.x;
  if (tid < BM) {
    int r = row0 + tid;
    bool ok = (r < n_e);
    tokS[tid] = ok ? tok_idx[e * B_ + r] : 0;
    gS[tid] = ok ? gates[e * B_ + r] : 0.0f;
  }
  __syncthreads();

  const __bf16* he = hidden + (size_t)e * B_ * H_;
#if !USE_TDM
  int mA[2], k8A[2];
#pragma unroll
  for (int i = 0; i < 2; ++i) {
    int c = tid + i * 256;            // 512 chunks = 128 rows x 4 octets
    mA[i] = c >> 2;
    k8A[i] = (c & 3) << 3;
  }
#endif
  int nB = tid & 127;
  int ksg = (tid >> 7) << 4;
  const float* bPtr = W2 + (size_t)e * H_ * T_ + (size_t)ksg * T_ + col0 + nB;

  int wave = tid >> 5, lane = tid & 31;
  int wm = (wave >> 2) * 64;
  int wn = (wave & 3) * 32;
  int lrow = lane & 15;
  int kh = (lane >> 4) << 3;

#if USE_TDM
  unsigned asOff = (unsigned)(uintptr_t)&As[0][0];  // LDS byte offset
#endif

  v8f acc[4][2] = {};

  for (int k0 = 0; k0 < H_; k0 += BK) {
    float bv[16];
#pragma unroll
    for (int kk = 0; kk < 16; ++kk) bv[kk] = bPtr[(size_t)(k0 + kk) * T_];
#if !USE_TDM
    bf16x8 av[2];
#pragma unroll
    for (int i = 0; i < 2; ++i)
      av[i] = *(const bf16x8*)(he + (size_t)(row0 + mA[i]) * H_ + k0 + k8A[i]);
#endif

    __syncthreads();  // previous iteration's fragment reads complete
#if USE_TDM
    if (wave == 0) {
      // DMA the 128x32-elem bf16 A tile straight into LDS (overlaps with the
      // VALU conversion work below); one issue per workgroup.
      tdm_load_tile_2d(asOff, he + (size_t)row0 * H_ + k0,
                       /*tile_w=*/BK, /*tile_h=*/BM, /*row_stride=*/H_,
                       /*tensor_w=*/H_, /*tensor_h=*/B_);
    }
#else
#pragma unroll
    for (int i = 0; i < 2; ++i) *(bf16x8*)(&As[mA[i]][k8A[i]]) = av[i];
#endif
    bf16x8 p0, p1;
#pragma unroll
    for (int kk = 0; kk < 8; ++kk) {
      p0[kk] = (__bf16)bv[kk];
      p1[kk] = (__bf16)bv[kk + 8];
    }
    *(bf16x8*)(&Bs[nB][ksg]) = p0;
    *(bf16x8*)(&Bs[nB][ksg + 8]) = p1;
#if USE_TDM
    if (wave == 0) __builtin_amdgcn_s_wait_tensorcnt(0);
#endif
    __syncthreads();

    v16bf bf0 = frag_ld(&Bs[wn + lrow][0], kh);
    v16bf bf1 = frag_ld(&Bs[wn + 16 + lrow][0], kh);
#pragma unroll
    for (int fm = 0; fm < 4; ++fm) {
      v16bf af = frag_ld(&As[wm + fm * 16 + lrow][0], kh);
      acc[fm][0] = __builtin_amdgcn_wmma_f32_16x16x32_bf16(
          false, af, false, bf0, (short)0, acc[fm][0], false, false);
      acc[fm][1] = __builtin_amdgcn_wmma_f32_16x16x32_bf16(
          false, af, false, bf1, (short)0, acc[fm][1], false, false);
    }
  }

  int mo = (lane >> 4) << 3;
  const float* b2e = b2 + e * T_;
#pragma unroll
  for (int fm = 0; fm < 4; ++fm) {
#pragma unroll
    for (int fn = 0; fn < 2; ++fn) {
#pragma unroll
      for (int i = 0; i < 8; ++i) {
        int rloc = wm + fm * 16 + mo + i;
        int rg = row0 + rloc;
        if (rg < n_e) {
          int t = col0 + wn + fn * 16 + lrow;
          float v = gS[rloc] * (acc[fm][fn][i] + b2e[t]);
          unsafeAtomicAdd(&out[(size_t)tokS[rloc] * T_ + t], v);
        }
      }
    }
  }
}

// ---------------------------------------------------------------------------
extern "C" void kernel_launch(void* const* d_in, const int* in_sizes, int n_in,
                              void* d_out, int out_size, void* d_ws, size_t ws_size,
                              hipStream_t stream) {
  const float* xs = (const float*)d_in[0];
  const float* Wg = (const float*)d_in[1];
  const float* bg = (const float*)d_in[2];
  const float* W1 = (const float*)d_in[3];
  const float* b1 = (const float*)d_in[4];
  const float* W2 = (const float*)d_in[5];
  const float* b2 = (const float*)d_in[6];

  float* out = (float*)d_out;               // [B, T]
  float* topv = out + (size_t)B_ * T_;      // [B, 2] tuple tail

  // Workspace layout (~134.5 MB):
  int* counts = (int*)d_ws;                 // [16] (8 used, padded)
  int* tok = counts + 16;                   // [E, B] compacted token indices
  float* gates = (float*)(tok + E_ * B_);   // [E, B] gate weights
  __bf16* hidden = (__bf16*)(gates + E_ * B_);  // [E, B, H] bf16

  zero_kernel<<<1024, 256, 0, stream>>>((float4*)out, counts);
  gate_topk_kernel<<<B_ / 8, 256, 0, stream>>>(xs, Wg, bg, topv, counts, tok, gates);

  dim3 g1(H_ / BN, B_ / BM, E_);
  expert_ffn1_kernel<<<g1, 256, 0, stream>>>(xs, W1, b1, counts, tok, hidden);

  dim3 g2(T_ / BN, B_ / BM, E_);
  expert_ffn2_kernel<<<g2, 256, 0, stream>>>(hidden, W2, b2, counts, tok, gates, out);
}
